// LSTMBlock_21964462751997
// MI455X (gfx1250) — matrix operand order repaired and checked by reading, not compiled, not run on hardware
//
#include <hip/hip_runtime.h>
#include <hip/hip_bf16.h>
#include <stdint.h>

// ---------------------------------------------------------------------------
// LSTM (B=32, T=512, IN=H=512) + ReLU for MI455X (gfx1250).
//
// Recurrence-latency-bound problem (69 GFLOP total, only ~110 MB of HBM
// traffic at 23.3 TB/s => HBM is ~5us; the 512 sequential steps dominate).
// One persistent scan kernel, 32 WGs x 256 threads (8 wave32). WG k owns
// hidden units [16k,16k+16). Per step: gates = [x_t ; h] (32x1024 bf16)
// @ W_fused^T via v_wmma_f32_16x16x32_bf16 (f32 accum), then LSTM cell.
//   - W slice (64 rows x 1024 K, bf16, 128 KB) staged ONCE into LDS by the
//     Tensor Data Mover (tensor_load_to_lds + s_wait_tensorcnt), then all
//     512 steps read B-fragments from LDS instead of re-pulling 128 KB/step
//     from L2.
//   - A-fragments (x_t, h) stream from global: x is read exactly once,
//     h (32 KB/step) is L2-hot; round-1 codegen showed these loads pipeline
//     under the WMMA chain.
//   - h exchanged via double-buffered global bf16 buffer + release/acquire
//     atomic-counter barrier (cluster HW barriers cap at 16 WGs and need
//     cluster dispatch; a NOP'd cluster barrier would be a silent race).
// Workspace: ctr + h[2][32][512]bf16 + Wfused[2048][1024]bf16 + x_bf16
//   = ~20.1 MB, all L2-resident (192 MB L2).
// ---------------------------------------------------------------------------

typedef __bf16 bf16;
typedef __attribute__((ext_vector_type(16))) __bf16 v16bf;
typedef __attribute__((ext_vector_type(8)))  __bf16 v8bf;
typedef __attribute__((ext_vector_type(8)))  float  v8f;
typedef __attribute__((ext_vector_type(4)))  unsigned int u32x4;
typedef __attribute__((ext_vector_type(8)))  int i32x8;
typedef __attribute__((ext_vector_type(4)))  int i32x4;

#define B_  32
#define T_  512
#define IN_ 512
#define H_  512
#define KF  1024          // fused K = IN + H
#define NG  2048          // 4*H gate rows
#define NWG 32            // scan workgroups
#define JW  16            // hidden units per WG

// ws layout (byte offsets)
#define CTR_OFF 0
#define HBF_OFF 256                       // 2*32*512*2   = 65536 B
#define WBF_OFF 65792                     // 2048*1024*2  = 4 MB
#define XBF_OFF 4260096                   // 32*512*512*2 = 16 MB

__global__ void init_kernel(unsigned int* ctr) {
  if (threadIdx.x == 0) *ctr = 0u;
}

__global__ void cvt_x_kernel(const float* __restrict__ x, bf16* __restrict__ xbf, int n) {
  int i = blockIdx.x * blockDim.x + threadIdx.x;
  int stride = gridDim.x * blockDim.x;
  for (; i < n; i += stride) xbf[i] = (bf16)x[i];
}

// Fused weight: row r (gate row), cols [0,512)=W_ih[r], [512,1024)=W_hh[r].
__global__ void fuse_w_kernel(const float* __restrict__ wih, const float* __restrict__ whh,
                              bf16* __restrict__ wbf) {
  int i = blockIdx.x * blockDim.x + threadIdx.x;
  int stride = gridDim.x * blockDim.x;
  const int n = NG * KF;
  for (; i < n; i += stride) {
    int r = i >> 10, c = i & 1023;
    float v = (c < IN_) ? wih[r * IN_ + c] : whh[r * H_ + (c - IN_)];
    wbf[i] = (bf16)v;
  }
}

__device__ __forceinline__ float sigmoidf_(float x) { return 1.f / (1.f + __expf(-x)); }
__device__ __forceinline__ float tanhf_(float x)    { return 1.f - 2.f / (__expf(2.f * x) + 1.f); }

// Issue one TDM 2-D tile load: 16 rows x 1024 bf16 from wbf row block -> LDS.
// D# bitfields per CDNA5 ISA.
__device__ __forceinline__ void tdm_load_wtile(const bf16* gsrc, unsigned lds_byte_addr) {
  unsigned long long ga = (unsigned long long)(uintptr_t)gsrc;
  u32x4 g0;
  g0.x = 1u;                                            // count=1, user desc
  g0.y = lds_byte_addr;                                 // lds_addr
  g0.z = (unsigned)(ga & 0xFFFFFFFFu);                  // global_addr[31:0]
  g0.w = (unsigned)((ga >> 32) & 0x1FFFFFFu)            // global_addr[56:32]
         | (2u << 30);                                  // type=2 ("image")
  i32x8 g1;
  g1[0] = (int)(1u << 16);                 // wg_mask=0 (no cluster), data_size=1 -> 2B
  g1[1] = (int)((KF & 0xFFFFu) << 16);     // [63:48] tensor_dim0 lo = 1024
  g1[2] = (int)(((unsigned)JW) << 16);     // [79:64] tdim0 hi=0; [95:80] tensor_dim1 lo=16
  g1[3] = (int)(((unsigned)KF) << 16);     // [111:96] tdim1 hi=0; [127:112] tile_dim0=1024
  g1[4] = (int)(JW);                       // [143:128] tile_dim1=16; [159:144] tile_dim2=0
  g1[5] = (int)(KF);                       // tensor_dim0_stride[31:0] = 1024
  g1[6] = 0;                               // stride hi, tensor_dim1_stride lo
  g1[7] = 0;
  i32x4 gz = {0, 0, 0, 0};
#if __clang_major__ >= 23
  i32x8 gz8 = {0, 0, 0, 0, 0, 0, 0, 0};
  __builtin_amdgcn_tensor_load_to_lds(g0, g1, gz, gz, gz8, 0);
#else
  __builtin_amdgcn_tensor_load_to_lds(g0, g1, gz, gz, 0);
#endif
}

__global__ void __launch_bounds__(256)
lstm_scan_kernel(const bf16* __restrict__ xbf, const bf16* __restrict__ wbf,
                 const float* __restrict__ b_ih, const float* __restrict__ b_hh,
                 bf16* __restrict__ hbf, unsigned int* __restrict__ ctr,
                 float* __restrict__ out) {
  extern __shared__ __align__(64) bf16 wlds[];  // [64][KF] bf16 = 128 KB (dynamic)
  __shared__ float glds[4][B_][JW];             // gate exchange, 8 KB

  const int tid  = threadIdx.x;
  const int wg   = blockIdx.x;                 // hidden slice [wg*16, wg*16+16)
  const int wave = tid >> 5, lane = tid & 31;
  const int half = lane >> 4, lr = lane & 15;
  const int q  = wave & 3;                     // gate: 0=i 1=f 2=g 3=o
  const int mt = wave >> 2;                    // batch tile (rows 0-15 / 16-31)
  const int m  = mt * 16 + lr;                 // A-fragment row = batch index
  const int n_row = q * H_ + wg * JW + lr;     // W row == B column for this lane
  const float bias = b_ih[n_row] + b_hh[n_row];

  // --- one-time: TDM-stage this WG's W slice (4 gate blocks) into LDS ------
  if (mt == 0) {                               // waves 0..3, wave id == q
    const bf16* gsrc = wbf + ((size_t)q * H_ + (size_t)wg * JW) * KF;
    unsigned lds_dst = (unsigned)(uintptr_t)wlds + (unsigned)(q * JW * KF * 2);
    tdm_load_wtile(gsrc, lds_dst);
    __builtin_amdgcn_s_wait_tensorcnt(0);
  }
  __syncthreads();

  const bf16* wrow = wlds + (size_t)(q * JW + lr) * KF;  // LDS B-row for this lane

  // per-thread cell state: 2 (b, n) pairs out of 32x16
  const int p0 = tid * 2;
  const int eb[2] = { p0 >> 4, (p0 + 1) >> 4 };
  const int en[2] = { p0 & 15, (p0 + 1) & 15 };
  float c_state[2] = { 0.f, 0.f };

  for (int t = 0; t < T_; ++t) {
    const int par = t & 1;
    v8f acc = { bias, bias, bias, bias, bias, bias, bias, bias };

    const bf16* arow_x = xbf + ((size_t)m * T_ + t) * IN_;
    const bf16* arow_h = hbf + (size_t)par * B_ * H_ + (size_t)m * H_;
    if (t + 1 < T_)
      __builtin_prefetch(xbf + ((size_t)m * T_ + t + 1) * IN_, 0, 0);

    // x contribution: K chunks 0..511
    #pragma unroll 8
    for (int kc = 0; kc < 16; ++kc) {
      const int kk = kc * 32;
      v16bf a = *(const v16bf*)(arow_x + kk + half * 16);
      v16bf b = *(const v16bf*)(wrow + kk + half * 16);
      acc = __builtin_amdgcn_wmma_f32_16x16x32_bf16(false, a, false, b, (short)0, acc, false, false);
    }
    // h contribution: K chunks 512..1023 (h_0 == 0 -> skip at t==0)
    if (t > 0) {
      #pragma unroll 8
      for (int kc = 0; kc < 16; ++kc) {
        const int kk = kc * 32;
        v16bf a = *(const v16bf*)(arow_h + kk + half * 16);
        v16bf b = *(const v16bf*)(wrow + IN_ + kk + half * 16);
        acc = __builtin_amdgcn_wmma_f32_16x16x32_bf16(false, a, false, b, (short)0, acc, false, false);
      }
    }

    // scatter C fragment (VGPR r: M=r (lanes 0-15) / M=r+8 (lanes 16-31), N=lr)
    #pragma unroll
    for (int r = 0; r < 8; ++r)
      glds[q][mt * 16 + r + 8 * half][lr] = acc[r];
    __syncthreads();

    // LSTM cell, 2 (b, n) pairs per thread
    #pragma unroll
    for (int e = 0; e < 2; ++e) {
      const int b = eb[e], n = en[e], j = wg * JW + n;
      float ig = sigmoidf_(glds[0][b][n]);
      float fg = sigmoidf_(glds[1][b][n]);
      float gg = tanhf_   (glds[2][b][n]);
      float og = sigmoidf_(glds[3][b][n]);
      float c = fg * c_state[e] + ig * gg;
      c_state[e] = c;
      float h = og * tanhf_(c);
      hbf[(size_t)(par ^ 1) * B_ * H_ + (size_t)b * H_ + j] = (bf16)h;
      out[((size_t)b * T_ + t) * H_ + j] = fmaxf(h, 0.f);
    }

    // inter-WG barrier: release h writes, arrive, spin until all 32 WGs done
    __threadfence();
    __syncthreads();
    if (t + 1 < T_) {
      if (tid == 0) {
        __hip_atomic_fetch_add(ctr, 1u, __ATOMIC_RELEASE, __HIP_MEMORY_SCOPE_AGENT);
        const unsigned int target = (unsigned int)NWG * (unsigned int)(t + 1);
        while (__hip_atomic_load(ctr, __ATOMIC_ACQUIRE, __HIP_MEMORY_SCOPE_AGENT) < target)
          __builtin_amdgcn_s_sleep(2);
      }
      __syncthreads();
      __threadfence();   // acquire side for all threads
    }
  }
}

extern "C" void kernel_launch(void* const* d_in, const int* in_sizes, int n_in,
                              void* d_out, int out_size, void* d_ws, size_t ws_size,
                              hipStream_t stream) {
  (void)in_sizes; (void)n_in; (void)out_size; (void)ws_size;
  const float* x    = (const float*)d_in[0];
  const float* W_ih = (const float*)d_in[1];
  const float* W_hh = (const float*)d_in[2];
  const float* b_ih = (const float*)d_in[3];
  const float* b_hh = (const float*)d_in[4];
  float* out = (float*)d_out;

  char* ws = (char*)d_ws;
  unsigned int* ctr = (unsigned int*)(ws + CTR_OFF);
  bf16* hbf = (bf16*)(ws + HBF_OFF);
  bf16* wbf = (bf16*)(ws + WBF_OFF);
  bf16* xbf = (bf16*)(ws + XBF_OFF);

  init_kernel<<<1, 64, 0, stream>>>(ctr);
  cvt_x_kernel<<<4096, 256, 0, stream>>>(x, xbf, B_ * T_ * IN_);
  fuse_w_kernel<<<2048, 256, 0, stream>>>(W_ih, W_hh, wbf);
  const size_t dyn_lds = (size_t)4 * JW * KF * sizeof(bf16);   // 128 KB
  lstm_scan_kernel<<<NWG, 256, dyn_lds, stream>>>(xbf, wbf, b_ih, b_hh, hbf, ctr, out);
}
